// VariantFModel_34892314313371
// MI455X (gfx1250) — matrix operand order repaired and checked by reading, not compiled, not run on hardware
//
#include <hip/hip_runtime.h>
#include <math.h>

// ---------------- model dimensions ----------------
constexpr int Bb = 4, Ss = 2048, Vv = 171, Dd = 896, ESs = 4;
constexpr int Ee = Ss / ESs;                 // 512 events
constexpr int NLn = 16;
constexpr int HDh = 384, TDt = 256, SDs = 256;
constexpr int GIg = 192, GHg = 4, GKd = GIg / GHg;   // 48
constexpr int SHs = 8, SKVs = 2, SHDs = SDs / SHs;   // 32
constexpr int BBUb = 448;
constexpr int EPPe = 8, Pp = Ee / EPPe;              // 64
constexpr int MEMm = 96, MHm = 8, MHDd = Dd / MHm;   // 112
constexpr int CDc = 256, CHc = 4, CHDd = CDc / CHc;  // 64

typedef __bf16 bf16;
typedef __attribute__((ext_vector_type(16))) bf16 bf16x16;
typedef __attribute__((ext_vector_type(8)))  bf16 bf16x8;
typedef __attribute__((ext_vector_type(8)))  float f32x8;

// CDNA5 async global->LDS path (ASYNCcnt), guarded so absence falls back cleanly.
#if defined(__has_builtin)
#  if __has_builtin(__builtin_amdgcn_global_load_async_to_lds_b128) && \
      __has_builtin(__builtin_amdgcn_s_wait_asynccnt) && defined(__AMDGCN__)
#    define HAVE_ASYNC_LDS 1
#  endif
#endif

#if defined(HAVE_ASYNC_LDS)
// builtin signature (from hipcc diagnostic): param0 = int4 pointer in global AS, non-const
typedef int v4i_async __attribute__((vector_size(16)));
typedef __attribute__((address_space(1))) v4i_async* gas4_ptr;
typedef __attribute__((address_space(3))) v4i_async* las4_ptr;
#endif

__device__ __forceinline__ void async_fence() {
#if defined(HAVE_ASYNC_LDS)
    __builtin_amdgcn_s_wait_asynccnt(0);
#endif
}

enum { A_NONE = 0, A_GELU = 1, A_SIG = 2 };

__device__ __forceinline__ float act_apply(float v, int act) {
    if (act == A_GELU) {
        float u = v * v * v;
        return 0.5f * v * (1.f + tanhf(0.7978845608028654f * (v + 0.044715f * u)));
    }
    if (act == A_SIG) return 1.f / (1.f + expf(-v));
    return v;
}

// convert 8 consecutive fp32 (16B-aligned LDS) to a bf16x8 fragment half
__device__ __forceinline__ bf16x8 cvt8(const float* p) {
    bf16x8 r;
    #pragma unroll
    for (int i = 0; i < 8; ++i) r[i] = (bf16)p[i];
    return r;
}

// ---------------- batched WMMA GEMM ----------------
// C[M,N] (+)= act(alpha*(A[M,K] @ B + bias)).
// transB=0: Bw is [K,N] row-major (weights);  transB=1: Bw is [N,K] row-major (Q@K^T).
// 256 threads = 8 wave32 waves; block tile 128x64, K-tile 32, double-buffered LDS,
// async global->LDS staging (ASYNCcnt) on interior tiles.
#define BMt 128
#define BNt 64
#define BKt 32
#define BKp (BKt + 4)   // row pad: keeps 16B alignment, spreads LDS banks

__global__ void __launch_bounds__(256)
k_gemm(const float* __restrict__ A, const float* __restrict__ Bw,
       const float* __restrict__ bias, float* __restrict__ C,
       int M, int N, int K, int lda, int ldb, int ldc,
       long long bsA, long long bsB, long long bsC,
       int transB, int act, int addC, float alpha)
{
    __shared__ alignas(16) float As[2][BMt][BKp];
    __shared__ alignas(16) float Bs[2][BNt][BKp];

    const int z = blockIdx.z;
    A  += (long long)z * bsA;
    Bw += (long long)z * bsB;
    C  += (long long)z * bsC;

    const int bm0 = blockIdx.y * BMt, bn0 = blockIdx.x * BNt;
    const int t = threadIdx.x, lane = t & 31, w = t >> 5;

    const bool aAligned = ((lda & 3) == 0);
    const bool bAligned = ((ldb & 3) == 0);
    const bool aFullM = (bm0 + BMt <= M);
    const bool bFullN = (bn0 + BNt <= N);

    // staging coordinates
    const int s_ar  = t >> 1;            // A: row 0..127, 16 floats each
    const int s_ac0 = (t & 1) * 16;
    const int s_bn  = t >> 2;            // B(transB=1): row 0..63, 8 floats along k
    const int s_bk0 = (t & 3) * 8;
    const int s_kk  = t >> 3;            // B(transB=0): k 0..31, 8 floats along n
    const int s_bn0 = (t & 7) * 8;

    auto stageA = [&](int buf, int k0) {
        int gm = bm0 + s_ar;
        const float* src = A + (long long)gm * lda + k0 + s_ac0;
        if (aFullM && (k0 + BKt <= K) && aAligned) {
#if defined(HAVE_ASYNC_LDS)
            #pragma unroll
            for (int j4 = 0; j4 < 4; ++j4)
                __builtin_amdgcn_global_load_async_to_lds_b128(
                    (gas4_ptr)(src + j4 * 4),
                    (las4_ptr)&As[buf][s_ar][s_ac0 + j4 * 4], 0, 0);
#else
            #pragma unroll
            for (int j4 = 0; j4 < 4; ++j4)
                *(float4*)&As[buf][s_ar][s_ac0 + j4 * 4] = *(const float4*)(src + j4 * 4);
#endif
        } else {
            bool ok = gm < M;
            #pragma unroll
            for (int j = 0; j < 16; ++j) {
                int gk = k0 + s_ac0 + j;
                As[buf][s_ar][s_ac0 + j] = (ok && gk < K) ? src[j] : 0.f;
            }
        }
    };

    auto stageB = [&](int buf, int k0) {
        if (transB) {
            int gn = bn0 + s_bn;
            const float* src = Bw + (long long)gn * ldb + k0 + s_bk0;
            if (bFullN && (k0 + BKt <= K) && bAligned) {
#if defined(HAVE_ASYNC_LDS)
                #pragma unroll
                for (int j4 = 0; j4 < 2; ++j4)
                    __builtin_amdgcn_global_load_async_to_lds_b128(
                        (gas4_ptr)(src + j4 * 4),
                        (las4_ptr)&Bs[buf][s_bn][s_bk0 + j4 * 4], 0, 0);
#else
                #pragma unroll
                for (int j4 = 0; j4 < 2; ++j4)
                    *(float4*)&Bs[buf][s_bn][s_bk0 + j4 * 4] = *(const float4*)(src + j4 * 4);
#endif
            } else {
                bool ok = gn < N;
                #pragma unroll
                for (int j = 0; j < 8; ++j) {
                    int gk = k0 + s_bk0 + j;
                    Bs[buf][s_bn][s_bk0 + j] = (ok && gk < K) ? src[j] : 0.f;
                }
            }
        } else {
            int gk = k0 + s_kk;
            const float* src = Bw + (long long)gk * ldb + bn0 + s_bn0;
            if (bFullN && (k0 + BKt <= K) && bAligned) {
                float4 v0 = *(const float4*)(src);
                float4 v1 = *(const float4*)(src + 4);
                Bs[buf][s_bn0 + 0][s_kk] = v0.x; Bs[buf][s_bn0 + 1][s_kk] = v0.y;
                Bs[buf][s_bn0 + 2][s_kk] = v0.z; Bs[buf][s_bn0 + 3][s_kk] = v0.w;
                Bs[buf][s_bn0 + 4][s_kk] = v1.x; Bs[buf][s_bn0 + 5][s_kk] = v1.y;
                Bs[buf][s_bn0 + 6][s_kk] = v1.z; Bs[buf][s_bn0 + 7][s_kk] = v1.w;
            } else {
                bool ok = gk < K;
                #pragma unroll
                for (int j = 0; j < 8; ++j) {
                    int gn = bn0 + s_bn0 + j;
                    Bs[buf][s_bn0 + j][s_kk] = (ok && gn < N) ? src[j] : 0.f;
                }
            }
        }
    };

    f32x8 acc[4];
    #pragma unroll
    for (int i = 0; i < 4; ++i)
        #pragma unroll
        for (int r = 0; r < 8; ++r) acc[i][r] = 0.f;

    // WMMA fragment coordinates (ISA 16-bit A / B layouts, wave32)
    const int am  = (lane & 15) + w * 16;
    const int klo = (lane & 16) ? 8 : 0;
    const int kb  = (lane & 16) ? 16 : 0;

    const int nk = (K + BKt - 1) / BKt;
    stageA(0, 0); stageB(0, 0);
    async_fence();
    __syncthreads();

    for (int kt = 0; kt < nk; ++kt) {
        const int cur = kt & 1;
        if (kt + 1 < nk) {               // prefetch next tile into the other buffer
            stageA(cur ^ 1, (kt + 1) * BKt);
            stageB(cur ^ 1, (kt + 1) * BKt);
        }

        // load ALL fragments first so ds_loads coalesce under one wait,
        // then fire the 4 WMMAs back-to-back (independent accumulators -> no hazard NOPs)
        const float* arow = &As[cur][am][0];
        bf16x8 a0 = cvt8(arow + klo);
        bf16x8 a1 = cvt8(arow + 16 + klo);
        bf16x16 afrag;
        #pragma unroll
        for (int i = 0; i < 8; ++i) { afrag[i] = a0[i]; afrag[i + 8] = a1[i]; }

        bf16x16 bfr[4];
        #pragma unroll
        for (int nt = 0; nt < 4; ++nt) {
            const float* brow = &Bs[cur][(lane & 15) + nt * 16][klo];
            bf16x8 b0 = cvt8(brow);
            bf16x8 b1 = cvt8(brow + 16);
            #pragma unroll
            for (int i = 0; i < 8; ++i) { bfr[nt][i] = b0[i]; bfr[nt][i + 8] = b1[i]; }
        }

        #pragma unroll
        for (int nt = 0; nt < 4; ++nt)
            acc[nt] = __builtin_amdgcn_wmma_f32_16x16x32_bf16(
                false, afrag, false, bfr[nt], (short)0, acc[nt], false, false);

        async_fence();       // next-tile async copies must land before next read
        __syncthreads();     // also orders manual ds stores + buffer reuse
    }

    // ---- epilogue: D layout VGPR r -> M = r + 8*(lane>=16), N = lane&15 ----
    const int ml = (lane >> 4) * 8;
    const int nn = lane & 15;
    for (int nt = 0; nt < 4; ++nt) {
        int gn = bn0 + nt * 16 + nn;
        if (gn >= N) continue;
        float bv = bias ? bias[gn] : 0.f;
        #pragma unroll
        for (int r = 0; r < 8; ++r) {
            int gm = bm0 + w * 16 + ml + r;
            if (gm >= M) continue;
            float v = act_apply(alpha * (acc[nt][r] + bv), act);
            long long idx = (long long)gm * ldc + gn;
            if (addC) C[idx] += v; else C[idx] = v;
        }
    }
}

// ---------------- layernorm (block=256, one row per block; in-place safe) ----------------
__global__ void k_layernorm(const float* __restrict__ x, const float* __restrict__ gw,
                            const float* __restrict__ bw, float* __restrict__ y, int dim)
{
    long long row = blockIdx.x;
    const float* xr = x + row * dim;
    float* yr = y + row * dim;
    __shared__ float red[256];
    int tid = threadIdx.x;
    float s = 0.f;
    for (int i = tid; i < dim; i += 256) s += xr[i];
    red[tid] = s; __syncthreads();
    for (int st = 128; st > 0; st >>= 1) { if (tid < st) red[tid] += red[tid + st]; __syncthreads(); }
    float mean = red[0] / dim; __syncthreads();
    float v = 0.f;
    for (int i = tid; i < dim; i += 256) { float d = xr[i] - mean; v += d * d; }
    red[tid] = v; __syncthreads();
    for (int st = 128; st > 0; st >>= 1) { if (tid < st) red[tid] += red[tid + st]; __syncthreads(); }
    float rstd = rsqrtf(red[0] / dim + 1e-5f);
    for (int i = tid; i < dim; i += 256) yr[i] = (xr[i] - mean) * rstd * gw[i] + bw[i];
}

// ---------------- embedding + time encoding ----------------
__global__ void k_embed(const int* __restrict__ tokens, const float* __restrict__ times,
                        const float* __restrict__ tok, const float* __restrict__ slot,
                        const float* __restrict__ epos, float* __restrict__ x)
{
    long long bs = blockIdx.x;                  // B*S rows
    int sidx = (int)(bs % Ss);
    int token = tokens[bs];
    float tval = times[bs];
    const float* trow = tok + (long long)token * Dd;
    const float* srow = slot + (long long)(sidx % ESs) * Dd;
    const float* erow = epos + (long long)(sidx / ESs) * Dd;
    float* xr = x + bs * Dd;
    const int half = Dd / 2;
    for (int d = threadIdx.x; d < Dd; d += blockDim.x) {
        float te;
        if (d < half) te = sinf(tval * expf(-logf(1200.f) * (float)d / half));
        else          te = cosf(tval * expf(-logf(1200.f) * (float)(d - half) / half));
        xr[d] = trow[d] + srow[d] + erow[d] + te;
    }
}

__global__ void k_ts(const float* __restrict__ times, float* __restrict__ ts)
{
    int i = blockIdx.x * blockDim.x + threadIdx.x;
    if (i >= Bb * Ee) return;
    int b = i / Ee, e = i % Ee;
    float cur = times[(long long)b * Ss + e * ESs + ESs - 1];
    float prev = (e == 0) ? cur : times[(long long)b * Ss + (e - 1) * ESs + ESs - 1];
    float d = cur - prev;
    ts[i] = d > 0.f ? d : 0.f;
}

// ---------------- rope on [BH, E, hd] ----------------
__global__ void k_rope(float* __restrict__ x, int E_, int hd)
{
    long long r = blockIdx.x;
    int e = (int)(r % E_);
    float* xr = x + r * hd;
    int half = hd / 2;
    for (int i = threadIdx.x; i < half; i += blockDim.x) {
        float inv = powf(10000.f, -(float)i / half);
        float ang = (float)e * inv, c = cosf(ang), s = sinf(ang);
        float x1 = xr[i], x2 = xr[i + half];
        xr[i] = x1 * c - x2 * s;
        xr[i + half] = x1 * s + x2 * c;
    }
}

// ---------------- head split / merge ----------------
__global__ void k_split_heads(const float* __restrict__ x, float* __restrict__ y,
                              int Bn, int E_, int Hin, int hd, int rep)
{
    int H = Hin * rep;
    long long total = (long long)Bn * H * E_ * hd;
    for (long long i = blockIdx.x * (long long)blockDim.x + threadIdx.x; i < total;
         i += (long long)gridDim.x * blockDim.x) {
        int d = (int)(i % hd); long long r = i / hd;
        int e = (int)(r % E_); r /= E_;
        int hh = (int)(r % H); int b = (int)(r / H);
        int hin = hh / rep;
        y[i] = x[((long long)b * E_ + e) * ((long long)Hin * hd) + (long long)hin * hd + d];
    }
}

__global__ void k_merge_heads(const float* __restrict__ y, float* __restrict__ x,
                              int Bn, int E_, int H, int hd)
{
    long long total = (long long)Bn * H * E_ * hd;
    for (long long i = blockIdx.x * (long long)blockDim.x + threadIdx.x; i < total;
         i += (long long)gridDim.x * blockDim.x) {
        int d = (int)(i % hd); long long r = i / hd;
        int e = (int)(r % E_); r /= E_;
        int hh = (int)(r % H); int b = (int)(r / H);
        x[((long long)b * E_ + e) * ((long long)H * hd) + (long long)hh * hd + d] = y[i];
    }
}

// ---------------- softmax (row per block; optional causal mask) ----------------
__global__ void k_softmax(float* __restrict__ sc, int Lq, int Lk, int causal)
{
    long long row = blockIdx.x;
    int qi = (int)(row % Lq);
    float* p = sc + row * Lk;
    __shared__ float red[256];
    int tid = threadIdx.x;
    float mx = -3.0e38f;
    for (int j = tid; j < Lk; j += 256) {
        float v = p[j];
        if (causal && j > qi) { v = -1.0e9f; p[j] = v; }
        mx = fmaxf(mx, v);
    }
    red[tid] = mx; __syncthreads();
    for (int st = 128; st > 0; st >>= 1) { if (tid < st) red[tid] = fmaxf(red[tid], red[tid + st]); __syncthreads(); }
    mx = red[0]; __syncthreads();
    float sum = 0.f;
    for (int j = tid; j < Lk; j += 256) { float e = expf(p[j] - mx); p[j] = e; sum += e; }
    red[tid] = sum; __syncthreads();
    for (int st = 128; st > 0; st >>= 1) { if (tid < st) red[tid] += red[tid + st]; __syncthreads(); }
    float inv = 1.f / red[0];
    for (int j = tid; j < Lk; j += 256) p[j] *= inv;
}

// ---------------- gated delta-net scan: one block per (b, head) ----------------
__global__ void __launch_bounds__(64)
k_gdn(const float* __restrict__ qg, const float* __restrict__ kg, const float* __restrict__ vg,
      const float* __restrict__ ag, const float* __restrict__ bg, float* __restrict__ outs)
{
    int bh = blockIdx.x;
    int b = bh / GHg, h = bh % GHg;
    __shared__ float kv[GKd], qv[GKd], vvv[GKd], ab[2];
    float Srow[GKd];
    int tid = threadIdx.x;
    #pragma unroll
    for (int i = 0; i < GKd; ++i) Srow[i] = 0.f;
    for (int e = 0; e < Ee; ++e) {
        long long base = (long long)b * Ee + e;
        if (tid < GKd) {
            kv[tid]  = kg[base * GIg + h * GKd + tid];
            qv[tid]  = qg[base * GIg + h * GKd + tid];
            vvv[tid] = vg[base * GIg + h * GKd + tid];
        }
        if (tid == 0) { ab[0] = ag[base * GHg + h]; ab[1] = bg[base * GHg + h]; }
        __syncthreads();
        if (tid < GKd) {
            float pred = 0.f;
            #pragma unroll
            for (int i = 0; i < GKd; ++i) pred += Srow[i] * kv[i];
            float coef = ab[1] * (vvv[tid] - pred);
            float at = ab[0], o = 0.f;
            #pragma unroll
            for (int i = 0; i < GKd; ++i) { Srow[i] = at * Srow[i] + coef * kv[i]; o += Srow[i] * qv[i]; }
            outs[base * GIg + h * GKd + tid] = o;
        }
        __syncthreads();
    }
}

__global__ void k_knorm(float* __restrict__ k)
{
    int i = blockIdx.x * blockDim.x + threadIdx.x;
    if (i >= Bb * Ee * GHg) return;
    float* p = k + (long long)i * GKd;
    float s = 1e-6f;
    for (int j = 0; j < GKd; ++j) s += p[j] * p[j];
    float r = rsqrtf(s);
    for (int j = 0; j < GKd; ++j) p[j] *= r;
}

// ---------------- CfC scan: one block per batch, 448 threads ----------------
__global__ void __launch_bounds__(BBUb)
k_cfc(const float* __restrict__ xin, const float* __restrict__ ts,
      const float* __restrict__ Wbb1, const float* __restrict__ bbb1,
      const float* __restrict__ Wbb2, const float* __restrict__ bbb2,
      const float* __restrict__ Wff1, const float* __restrict__ bff1,
      const float* __restrict__ Wff2, const float* __restrict__ bff2,
      const float* __restrict__ Wta, const float* __restrict__ bta,
      const float* __restrict__ Wtb, const float* __restrict__ btb,
      float* __restrict__ outs)
{
    int b = blockIdx.x;
    __shared__ float xh[2 * TDt];
    __shared__ float bb[BBUb], bb2[BBUb];
    int tid = threadIdx.x;
    if (tid < TDt) xh[TDt + tid] = 0.f;   // h0 = 0
    __syncthreads();
    for (int e = 0; e < Ee; ++e) {
        if (tid < TDt) xh[tid] = xin[((long long)b * Ee + e) * TDt + tid];
        __syncthreads();
        {   // bb = tanh([xt,h] @ Wbb1 + b)
            float acc = bbb1[tid];
            for (int i = 0; i < 2 * TDt; ++i) acc += xh[i] * Wbb1[(long long)i * BBUb + tid];
            bb[tid] = tanhf(acc);
        }
        __syncthreads();
        {   // bb2 = tanh(bb @ Wbb2 + b)
            float acc = bbb2[tid];
            for (int i = 0; i < BBUb; ++i) acc += bb[i] * Wbb2[(long long)i * BBUb + tid];
            bb2[tid] = tanhf(acc);
        }
        __syncthreads();
        if (tid < TDt) {
            float a1 = bff1[tid], a2 = bff2[tid], aa = bta[tid], atb = btb[tid];
            for (int i = 0; i < BBUb; ++i) {
                float bv = bb2[i];
                a1  += bv * Wff1[(long long)i * TDt + tid];
                a2  += bv * Wff2[(long long)i * TDt + tid];
                aa  += bv * Wta[(long long)i * TDt + tid];
                atb += bv * Wtb[(long long)i * TDt + tid];
            }
            float f1 = tanhf(a1), f2 = tanhf(a2);
            float ta = (aa > 20.f) ? aa : log1pf(expf(aa));     // softplus
            float tt = ts[b * Ee + e];
            float ti = 1.f / (1.f + expf(ta * tt - atb));        // sigmoid(-ta*tt + tb)
            float hn = f1 * (1.f - ti) + ti * f2;
            outs[((long long)b * Ee + e) * TDt + tid] = hn;
            xh[TDt + tid] = hn;
        }
        __syncthreads();
    }
}

// ---------------- small elementwise helpers ----------------
__global__ void k_copy_cols(const float* __restrict__ src, float* __restrict__ dst,
                            long long rows, int cols, int sld, int dld)
{
    long long total = rows * cols;
    for (long long i = blockIdx.x * (long long)blockDim.x + threadIdx.x; i < total;
         i += (long long)gridDim.x * blockDim.x) {
        long long r = i / cols; int c = (int)(i % cols);
        dst[r * dld + c] = src[r * sld + c];
    }
}

__global__ void k_ew_gate(float* __restrict__ ht, const float* __restrict__ gate,
                          const float* __restrict__ delta, long long n)
{
    for (long long i = blockIdx.x * (long long)blockDim.x + threadIdx.x; i < n;
         i += (long long)gridDim.x * blockDim.x)
        ht[i] += gate[i] * delta[i];
}

__global__ void k_add_rowvec(float* __restrict__ out, const float* __restrict__ x,
                             const float* __restrict__ vec, long long rows, int dim)
{
    long long n = rows * dim;
    for (long long i = blockIdx.x * (long long)blockDim.x + threadIdx.x; i < n;
         i += (long long)gridDim.x * blockDim.x)
        out[i] = x[i] + vec[i % dim];
}

__global__ void k_meanpool(const float* __restrict__ m, float* __restrict__ y)
{
    long long total = (long long)Bb * Pp * Dd;
    for (long long i = blockIdx.x * (long long)blockDim.x + threadIdx.x; i < total;
         i += (long long)gridDim.x * blockDim.x) {
        int d = (int)(i % Dd); long long r = i / Dd;
        int pp = (int)(r % Pp); int b = (int)(r / Pp);
        float s = 0.f;
        for (int j = 0; j < EPPe; ++j)
            s += m[((long long)b * Ee + pp * EPPe + j) * Dd + d];
        y[i] = s / EPPe;
    }
}

__global__ void k_add_repeat(float* __restrict__ m, const float* __restrict__ y)
{
    long long total = (long long)Bb * Ee * Dd;
    for (long long i = blockIdx.x * (long long)blockDim.x + threadIdx.x; i < total;
         i += (long long)gridDim.x * blockDim.x) {
        int d = (int)(i % Dd); long long r = i / Dd;
        int e = (int)(r % Ee); int b = (int)(r / Ee);
        m[i] += y[((long long)b * Pp + e / EPPe) * Dd + d];
    }
}

// ================= host side =================
struct Cursor { void* const* p; int i; const float* f() { return (const float*)p[i++]; } };
struct Lin { const float* W; const float* b; };
struct LNp { const float* b; const float* g; };
static Lin getW(Cursor& c)  { Lin l; l.W = c.f(); l.b = c.f(); return l; }  // {'W','b'} sorted: W,b
static LNp getLN(Cursor& c) { LNp l; l.b = c.f(); l.g = c.f(); return l; }  // {'g','b'} sorted: b,g

struct LayerP { Lin bb1, bb2, ff1, ff2, ga, gb, gk, go, gq, gv; LNp hn;
                Lin sk; LNp sn; Lin so, sq, sv, ta, tb, ti; LNp tn; Lin to; };
struct CrossP { Lin g_l; LNp g_ln; Lin hk, ho, hq, ht_l; LNp ht_ln; Lin hv, sh_l; LNp sh_ln;
                Lin tk, to, tq, tv; };
struct HeadP  { Lin l1, l2; LNp ln; };

struct Arena {
    char* p; size_t off;
    float* get(size_t n) {
        float* r = (float*)(p + off);
        off += ((n * sizeof(float)) + 255) & ~(size_t)255;
        return r;
    }
};

static void gemm(hipStream_t st, const float* A, const float* Bw, const float* bias, float* C,
                 int M, int N, int K, int lda, int ldb, int ldc,
                 int batch, long long bsA, long long bsB, long long bsC,
                 int transB, int act, int addC, float alpha)
{
    dim3 g((N + BNt - 1) / BNt, (M + BMt - 1) / BMt, batch);
    k_gemm<<<g, 256, 0, st>>>(A, Bw, bias, C, M, N, K, lda, ldb, ldc,
                              bsA, bsB, bsC, transB, act, addC, alpha);
}

static void lnorm(hipStream_t st, const float* x, const LNp& p, float* y, long long rows, int dim)
{
    k_layernorm<<<dim3((unsigned)rows), 256, 0, st>>>(x, p.g, p.b, y, dim);
}

// q,k,v: [BH, Lq|Lk, hd] -> out [BH, Lq, hd] (scores is scratch [BH, Lq, Lk])
static void attn(hipStream_t st, const float* q, const float* k, const float* v,
                 float* scores, float* out, int BH, int Lq, int Lk, int hd, int causal)
{
    float sc = 1.f / sqrtf((float)hd);
    gemm(st, q, k, nullptr, scores, Lq, Lk, hd, hd, hd, Lk,
         BH, (long long)Lq * hd, (long long)Lk * hd, (long long)Lq * Lk,
         1, A_NONE, 0, sc);
    k_softmax<<<dim3((unsigned)((long long)BH * Lq)), 256, 0, st>>>(scores, Lq, Lk, causal);
    gemm(st, scores, v, nullptr, out, Lq, hd, Lk, Lk, hd, hd,
         BH, (long long)Lq * Lk, (long long)Lk * hd, (long long)Lq * hd,
         0, A_NONE, 0, 1.f);
}

extern "C" void kernel_launch(void* const* d_in, const int* in_sizes, int n_in,
                              void* d_out, int out_size, void* d_ws, size_t ws_size,
                              hipStream_t stream)
{
    (void)in_sizes; (void)n_in; (void)out_size; (void)ws_size;
    const int*   tokens = (const int*)d_in[0];
    const float* times  = (const float*)d_in[1];
    Cursor c{d_in, 2};

    // ---- params, jax pytree (sorted-key, depth-first) order ----
    CrossP cr[NLn / 2];
    for (int i = 0; i < NLn / 2; ++i) { CrossP& q = cr[i];
        q.g_l = getW(c); q.g_ln = getLN(c); q.hk = getW(c); q.ho = getW(c); q.hq = getW(c);
        q.ht_l = getW(c); q.ht_ln = getLN(c); q.hv = getW(c); q.sh_l = getW(c); q.sh_ln = getLN(c);
        q.tk = getW(c); q.to = getW(c); q.tq = getW(c); q.tv = getW(c);
    }
    Lin e2t = getW(c);
    const float* epos = c.f();
    LNp fln = getLN(c);
    Lin h_in = getW(c); LNp h_in_ln = getLN(c);
    HeadP hp[ESs];
    for (int i = 0; i < ESs; ++i) { hp[i].l1 = getW(c); hp[i].l2 = getW(c); hp[i].ln = getLN(c); }
    LayerP L[NLn];
    for (int i = 0; i < NLn; ++i) { LayerP& l = L[i];
        l.bb1 = getW(c); l.bb2 = getW(c); l.ff1 = getW(c); l.ff2 = getW(c);
        l.ga = getW(c); l.gb = getW(c); l.gk = getW(c); l.go = getW(c); l.gq = getW(c); l.gv = getW(c);
        l.hn = getLN(c);
        l.sk = getW(c); l.sn = getLN(c); l.so = getW(c); l.sq = getW(c); l.sv = getW(c);
        l.ta = getW(c); l.tb = getW(c); l.ti = getW(c); l.tn = getLN(c); l.to = getW(c);
    }
    Lin m_l1 = getW(c), m_l2 = getW(c); LNp m_ln = getLN(c), m_ln2 = getLN(c);
    const float* memE = c.f();
    Lin mk = getW(c); LNp mln = getLN(c); Lin mo = getW(c), mq = getW(c), mv = getW(c);
    Lin p2e = getW(c), pack1 = getW(c), pack2 = getW(c);
    LNp pack_ln = getLN(c), pack_ln2 = getLN(c);
    Lin pf1 = getW(c), pf2 = getW(c), pkL = getW(c);
    LNp pln1 = getLN(c), pln2 = getLN(c);
    Lin poL = getW(c), pqL = getW(c), psum = getW(c), pvL = getW(c);
    Lin s_in = getW(c); LNp s_in_ln = getLN(c);
    const float* slot = c.f(); const float* slotdec = c.f();
    Lin t_in = getW(c); LNp t_in_ln = getLN(c);
    const float* tokE = c.f();

    // ---- workspace ----
    const long long BE  = (long long)Bb * Ee;     // 2048 rows
    const long long BED = BE * Dd;
    Arena ar{(char*)d_ws, 0};
    float* xf     = ar.get((size_t)Bb * Ss * Dd);
    float* ev     = ar.get(BED);
    float* hB     = ar.get(BE * HDh);
    float* tB     = ar.get(BE * TDt);
    float* sB     = ar.get(BE * SDs);
    float* tsb    = ar.get((size_t)Bb * Ee);
    float* mB     = ar.get(BED);
    float* scores = ar.get((size_t)Bb * SHs * Ee * Ee);
    float* g[8];
    for (int i = 0; i < 8; ++i) g[i] = ar.get(BED);

    const dim3 G1(2048), B1(256);

    // ---- embedding, event times ----
    k_embed<<<Bb * Ss, 256, 0, stream>>>(tokens, times, tokE, slot, epos, xf);
    k_ts<<<(Bb * Ee + 255) / 256, 256, 0, stream>>>(times, tsb);

    // ---- pack to event vectors ----
    lnorm(stream, xf, pack_ln, xf, BE, ESs * Dd);
    gemm(stream, xf, pack1.W, pack1.b, g[0], (int)BE, Dd, ESs * Dd, ESs * Dd, Dd, Dd,
         1, 0, 0, 0, 0, A_GELU, 0, 1.f);
    gemm(stream, g[0], pack2.W, pack2.b, ev, (int)BE, Dd, Dd, Dd, Dd, Dd,
         1, 0, 0, 0, 0, A_NONE, 0, 1.f);
    lnorm(stream, ev, pack_ln2, ev, BE, Dd);
    gemm(stream, ev, h_in.W, h_in.b, hB, (int)BE, HDh, Dd, Dd, HDh, HDh, 1, 0, 0, 0, 0, A_NONE, 0, 1.f);
    lnorm(stream, hB, h_in_ln, hB, BE, HDh);
    gemm(stream, ev, t_in.W, t_in.b, tB, (int)BE, TDt, Dd, Dd, TDt, TDt, 1, 0, 0, 0, 0, A_NONE, 0, 1.f);
    lnorm(stream, tB, t_in_ln, tB, BE, TDt);
    gemm(stream, ev, s_in.W, s_in.b, sB, (int)BE, SDs, Dd, Dd, SDs, SDs, 1, 0, 0, 0, 0, A_NONE, 0, 1.f);
    lnorm(stream, sB, s_in_ln, sB, BE, SDs);

    // ---- layers ----
    for (int li = 0; li < NLn; ++li) {
        const LayerP& l = L[li];
        // --- GDN branch: h += gdn(ln(h)) ---
        lnorm(stream, hB, l.hn, g[0], BE, HDh);
        gemm(stream, g[0], l.gq.W, l.gq.b, g[1], (int)BE, GIg, HDh, HDh, GIg, GIg, 1, 0, 0, 0, 0, A_NONE, 0, 1.f);
        gemm(stream, g[0], l.gk.W, l.gk.b, g[2], (int)BE, GIg, HDh, HDh, GIg, GIg, 1, 0, 0, 0, 0, A_NONE, 0, 1.f);
        gemm(stream, g[0], l.gv.W, l.gv.b, g[3], (int)BE, GIg, HDh, HDh, GIg, GIg, 1, 0, 0, 0, 0, A_NONE, 0, 1.f);
        gemm(stream, g[0], l.ga.W, l.ga.b, g[4], (int)BE, GHg, HDh, HDh, GHg, GHg, 1, 0, 0, 0, 0, A_SIG, 0, 1.f);
        gemm(stream, g[0], l.gb.W, l.gb.b, g[5], (int)BE, GHg, HDh, HDh, GHg, GHg, 1, 0, 0, 0, 0, A_SIG, 0, 1.f);
        k_knorm<<<(Bb * Ee * GHg + 255) / 256, 256, 0, stream>>>(g[2]);
        k_gdn<<<Bb * GHg, 64, 0, stream>>>(g[1], g[2], g[3], g[4], g[5], g[6]);
        gemm(stream, g[6], l.go.W, l.go.b, hB, (int)BE, HDh, GIg, GIg, HDh, HDh, 1, 0, 0, 0, 0, A_NONE, 1, 1.f);

        // --- CfC branch: t += lin(scan(lin(ln(t))), to) ---
        lnorm(stream, tB, l.tn, g[0], BE, TDt);
        gemm(stream, g[0], l.ti.W, l.ti.b, g[1], (int)BE, TDt, TDt, TDt, TDt, TDt, 1, 0, 0, 0, 0, A_NONE, 0, 1.f);
        k_cfc<<<Bb, BBUb, 0, stream>>>(g[1], tsb, l.bb1.W, l.bb1.b, l.bb2.W, l.bb2.b,
                                       l.ff1.W, l.ff1.b, l.ff2.W, l.ff2.b,
                                       l.ta.W, l.ta.b, l.tb.W, l.tb.b, g[2]);
        gemm(stream, g[2], l.to.W, l.to.b, tB, (int)BE, TDt, TDt, TDt, TDt, TDt, 1, 0, 0, 0, 0, A_NONE, 1, 1.f);

        // --- GQA branch: s += gqa(ln(s)) ---
        lnorm(stream, sB, l.sn, g[0], BE, SDs);
        gemm(stream, g[0], l.sq.W, l.sq.b, g[1], (int)BE, SHs * SHDs, SDs, SDs, SHs * SHDs, SHs * SHDs, 1, 0, 0, 0, 0, A_NONE, 0, 1.f);
        gemm(stream, g[0], l.sk.W, l.sk.b, g[2], (int)BE, SKVs * SHDs, SDs, SDs, SKVs * SHDs, SKVs * SHDs, 1, 0, 0, 0, 0, A_NONE, 0, 1.f);
        gemm(stream, g[0], l.sv.W, l.sv.b, g[3], (int)BE, SKVs * SHDs, SDs, SDs, SKVs * SHDs, SKVs * SHDs, 1, 0, 0, 0, 0, A_NONE, 0, 1.f);
        k_split_heads<<<G1, B1, 0, stream>>>(g[1], g[4], Bb, Ee, SHs, SHDs, 1);
        k_split_heads<<<G1, B1, 0, stream>>>(g[2], g[5], Bb, Ee, SKVs, SHDs, SHs / SKVs);
        k_split_heads<<<G1, B1, 0, stream>>>(g[3], g[6], Bb, Ee, SKVs, SHDs, SHs / SKVs);
        k_rope<<<Bb * SHs * Ee, 64, 0, stream>>>(g[4], Ee, SHDs);
        k_rope<<<Bb * SHs * Ee, 64, 0, stream>>>(g[5], Ee, SHDs);
        attn(stream, g[4], g[5], g[6], scores, g[7], Bb * SHs, Ee, Ee, SHDs, 1);
        k_merge_heads<<<G1, B1, 0, stream>>>(g[7], g[0], Bb, Ee, SHs, SHDs);
        gemm(stream, g[0], l.so.W, l.so.b, sB, (int)BE, SDs, SDs, SDs, SDs, SDs, 1, 0, 0, 0, 0, A_NONE, 1, 1.f);

        // --- cross mixing every second layer ---
        if ((li + 1) % 2 == 0) {
            const CrossP& q = cr[li / 2];
            // h += xattn(h, t)
            gemm(stream, hB, q.hq.W, q.hq.b, g[0], (int)BE, CDc, HDh, HDh, CDc, CDc, 1, 0, 0, 0, 0, A_NONE, 0, 1.f);
            gemm(stream, tB, q.tk.W, q.tk.b, g[1], (int)BE, CDc, TDt, TDt, CDc, CDc, 1, 0, 0, 0, 0, A_NONE, 0, 1.f);
            gemm(stream, tB, q.tv.W, q.tv.b, g[2], (int)BE, CDc, TDt, TDt, CDc, CDc, 1, 0, 0, 0, 0, A_NONE, 0, 1.f);
            k_split_heads<<<G1, B1, 0, stream>>>(g[0], g[3], Bb, Ee, CHc, CHDd, 1);
            k_split_heads<<<G1, B1, 0, stream>>>(g[1], g[4], Bb, Ee, CHc, CHDd, 1);
            k_split_heads<<<G1, B1, 0, stream>>>(g[2], g[5], Bb, Ee, CHc, CHDd, 1);
            attn(stream, g[3], g[4], g[5], scores, g[6], Bb * CHc, Ee, Ee, CHDd, 1);
            k_merge_heads<<<G1, B1, 0, stream>>>(g[6], g[0], Bb, Ee, CHc, CHDd);
            gemm(stream, g[0], q.ho.W, q.ho.b, hB, (int)BE, HDh, CDc, CDc, HDh, HDh, 1, 0, 0, 0, 0, A_NONE, 1, 1.f);
            // t += xattn(t, updated h)
            gemm(stream, tB, q.tq.W, q.tq.b, g[0], (int)BE, CDc, TDt, TDt, CDc, CDc, 1, 0, 0, 0, 0, A_NONE, 0, 1.f);
            gemm(stream, hB, q.hk.W, q.hk.b, g[1], (int)BE, CDc, HDh, HDh, CDc, CDc, 1, 0, 0, 0, 0, A_NONE, 0, 1.f);
            gemm(stream, hB, q.hv.W, q.hv.b, g[2], (int)BE, CDc, HDh, HDh, CDc, CDc, 1, 0, 0, 0, 0, A_NONE, 0, 1.f);
            k_split_heads<<<G1, B1, 0, stream>>>(g[0], g[3], Bb, Ee, CHc, CHDd, 1);
            k_split_heads<<<G1, B1, 0, stream>>>(g[1], g[4], Bb, Ee, CHc, CHDd, 1);
            k_split_heads<<<G1, B1, 0, stream>>>(g[2], g[5], Bb, Ee, CHc, CHDd, 1);
            attn(stream, g[3], g[4], g[5], scores, g[6], Bb * CHc, Ee, Ee, CHDd, 1);
            k_merge_heads<<<G1, B1, 0, stream>>>(g[6], g[0], Bb, Ee, CHc, CHDd);
            gemm(stream, g[0], q.to.W, q.to.b, tB, (int)BE, TDt, CDc, CDc, TDt, TDt, 1, 0, 0, 0, 0, A_NONE, 1, 1.f);
            // ht = [h, t];  s += gelu(lin(ln(ht)))
            const int HT = HDh + TDt;
            k_copy_cols<<<G1, B1, 0, stream>>>(hB, g[0], BE, HDh, HDh, HT);
            k_copy_cols<<<G1, B1, 0, stream>>>(tB, g[0] + HDh, BE, TDt, TDt, HT);
            lnorm(stream, g[0], q.ht_ln, g[1], BE, HT);
            gemm(stream, g[1], q.ht_l.W, q.ht_l.b, sB, (int)BE, SDs, HT, HT, SDs, SDs, 1, 0, 0, 0, 0, A_GELU, 1, 1.f);
            // delta / gate from updated s; ht += gate*delta; split back
            lnorm(stream, sB, q.sh_ln, g[1], BE, SDs);
            gemm(stream, g[1], q.sh_l.W, q.sh_l.b, g[2], (int)BE, HT, SDs, SDs, HT, HT, 1, 0, 0, 0, 0, A_GELU, 0, 1.f);
            lnorm(stream, sB, q.g_ln, g[1], BE, SDs);
            gemm(stream, g[1], q.g_l.W, q.g_l.b, g[3], (int)BE, HT, SDs, SDs, HT, HT, 1, 0, 0, 0, 0, A_SIG, 0, 1.f);
            k_ew_gate<<<G1, B1, 0, stream>>>(g[0], g[3], g[2], BE * HT);
            k_copy_cols<<<G1, B1, 0, stream>>>(g[0], hB, BE, HDh, HT, HDh);
            k_copy_cols<<<G1, B1, 0, stream>>>(g[0] + HDh, tB, BE, TDt, HT, TDt);
        }
    }

    // ---- merge streams ----
    k_copy_cols<<<G1, B1, 0, stream>>>(hB, g[0], BE, HDh, HDh, Dd);
    k_copy_cols<<<G1, B1, 0, stream>>>(tB, g[0] + HDh, BE, TDt, TDt, Dd);
    k_copy_cols<<<G1, B1, 0, stream>>>(sB, g[0] + HDh + TDt, BE, SDs, SDs, Dd);
    lnorm(stream, g[0], m_ln, g[0], BE, Dd);
    gemm(stream, g[0], m_l1.W, m_l1.b, g[1], (int)BE, Dd, Dd, Dd, Dd, Dd, 1, 0, 0, 0, 0, A_GELU, 0, 1.f);
    gemm(stream, g[1], m_l2.W, m_l2.b, mB, (int)BE, Dd, Dd, Dd, Dd, Dd, 1, 0, 0, 0, 0, A_NONE, 0, 1.f);
    lnorm(stream, mB, m_ln2, mB, BE, Dd);

    // ---- pooled path ----
    const int BP = Bb * Pp;  // 256 rows
    k_meanpool<<<G1, B1, 0, stream>>>(mB, g[0]);
    gemm(stream, g[0], psum.W, psum.b, g[1], BP, Dd, Dd, Dd, Dd, Dd, 1, 0, 0, 0, 0, A_NONE, 0, 1.f);   // ph
    lnorm(stream, g[1], pln1, g[2], BP, Dd);
    gemm(stream, g[2], pqL.W, pqL.b, g[3], BP, Dd, Dd, Dd, Dd, Dd, 1, 0, 0, 0, 0, A_NONE, 0, 1.f);
    gemm(stream, g[2], pkL.W, pkL.b, g[4], BP, Dd, Dd, Dd, Dd, Dd, 1, 0, 0, 0, 0, A_NONE, 0, 1.f);
    gemm(stream, g[2], pvL.W, pvL.b, g[5], BP, Dd, Dd, Dd, Dd, Dd, 1, 0, 0, 0, 0, A_NONE, 0, 1.f);
    k_split_heads<<<G1, B1, 0, stream>>>(g[3], g[6], Bb, Pp, MHm, MHDd, 1);
    k_split_heads<<<G1, B1, 0, stream>>>(g[4], g[7], Bb, Pp, MHm, MHDd, 1);
    k_split_heads<<<G1, B1, 0, stream>>>(g[5], g[0], Bb, Pp, MHm, MHDd, 1);
    attn(stream, g[6], g[7], g[0], scores, g[2], Bb * MHm, Pp, Pp, MHDd, 1);
    k_merge_heads<<<G1, B1, 0, stream>>>(g[2], g[3], Bb, Pp, MHm, MHDd);
    gemm(stream, g[3], poL.W, poL.b, g[1], BP, Dd, Dd, Dd, Dd, Dd, 1, 0, 0, 0, 0, A_NONE, 1, 1.f);     // ph += attn
    lnorm(stream, g[1], pln2, g[2], BP, Dd);
    gemm(stream, g[2], pf1.W, pf1.b, g[3], BP, 2 * Dd, Dd, Dd, 2 * Dd, 2 * Dd, 1, 0, 0, 0, 0, A_GELU, 0, 1.f);
    gemm(stream, g[3], pf2.W, pf2.b, g[1], BP, Dd, 2 * Dd, 2 * Dd, Dd, Dd, 1, 0, 0, 0, 0, A_NONE, 1, 1.f);

    // ---- memory attention ----
    lnorm(stream, g[1], mln, g[2], BP, Dd);
    gemm(stream, g[2], mq.W, mq.b, g[3], BP, Dd, Dd, Dd, Dd, Dd, 1, 0, 0, 0, 0, A_NONE, 0, 1.f);
    gemm(stream, memE, mk.W, mk.b, g[4], MEMm, Dd, Dd, Dd, Dd, Dd, 1, 0, 0, 0, 0, A_NONE, 0, 1.f);
    gemm(stream, memE, mv.W, mv.b, g[5], MEMm, Dd, Dd, Dd, Dd, Dd, 1, 0, 0, 0, 0, A_NONE, 0, 1.f);
    k_split_heads<<<G1, B1, 0, stream>>>(g[3], g[6], Bb, Pp, MHm, MHDd, 1);
    k_split_heads<<<G1, B1, 0, stream>>>(g[4], g[7], 1, MEMm, MHm, MHDd, 1);
    k_split_heads<<<G1, B1, 0, stream>>>(g[5], g[0], 1, MEMm, MHm, MHDd, 1);
    for (int b = 0; b < Bb; ++b)
        attn(stream, g[6] + (long long)b * MHm * Pp * MHDd, g[7], g[0], scores,
             g[2] + (long long)b * MHm * Pp * MHDd, MHm, Pp, MEMm, MHDd, 0);
    k_merge_heads<<<G1, B1, 0, stream>>>(g[2], g[3], Bb, Pp, MHm, MHDd);
    gemm(stream, g[3], mo.W, mo.b, g[1], BP, Dd, Dd, Dd, Dd, Dd, 1, 0, 0, 0, 0, A_NONE, 1, 1.f);       // ph += mem
    gemm(stream, g[1], p2e.W, p2e.b, g[2], BP, Dd, Dd, Dd, Dd, Dd, 1, 0, 0, 0, 0, A_NONE, 0, 1.f);
    k_add_repeat<<<G1, B1, 0, stream>>>(mB, g[2]);

    // ---- decode heads ----
    lnorm(stream, mB, fln, g[0], BE, Dd);
    gemm(stream, g[0], e2t.W, e2t.b, g[1], (int)BE, Dd, Dd, Dd, Dd, Dd, 1, 0, 0, 0, 0, A_NONE, 0, 1.f); // xe
    const float SCALE = 1.f / sqrtf((float)Dd);
    float* out = (float*)d_out;
    for (int si = 0; si < ESs; ++si) {
        k_add_rowvec<<<G1, B1, 0, stream>>>(g[2], g[1], slotdec + (long long)si * Dd, BE, Dd);
        lnorm(stream, g[2], hp[si].ln, g[2], BE, Dd);
        gemm(stream, g[2], hp[si].l1.W, hp[si].l1.b, g[3], (int)BE, Dd, Dd, Dd, Dd, Dd,
             1, 0, 0, 0, 0, A_GELU, 0, 1.f);
        // logits for slot si interleave into [B, E, ES, V] == [B, S, V]
        gemm(stream, g[3], hp[si].l2.W, hp[si].l2.b, out + (long long)si * Vv,
             (int)BE, Vv, Dd, Dd, Vv, ESs * Vv, 1, 0, 0, 0, 0, A_NONE, 0, SCALE);
    }
}
